// GRU_69818988364526
// MI455X (gfx1250) — matrix operand order repaired and checked by reading, not compiled, not run on hardware
//
#include <hip/hip_runtime.h>
#include <hip/hip_bf16.h>
#include <stdint.h>

// GRU: T=512 steps of  z,r = sigmoid([h,x]W^T+b);  h' = (1-z)h + z tanh([r*h,x]Wh^T+bh)
// Latency-bound recurrence (151 MFLOP/step). Persistent 16-WG kernel, weights
// resident in LDS (144KB/WGP of 320KB), bf16 WMMA 16x16x32 w/ f32 accum,
// 2 grid barriers per step (reset applies BEFORE the candidate GEMM).
// Round 2: branch-free, fully-unrolled K loops -> immediate-offset addressing,
// load/WMMA overlap instead of s_wait 0x0 stalls.

#define T_   512
#define B_   64
#define I_   256
#define L_   512
#define HX_  768
#define NWG  16
#define KT_TOT 24           // 768/32 K-tiles of hx
#define KT_H   16           // 512/32 K-tiles of the h part
#define KT_X   8            // 256/32 K-tiles of the x part
#define FRAG_EL 512         // one 16x32 bf16 fragment = 32 lanes * 16 elems
#define SLOT_BYTES 24576    // 24 frags * 512 el * 2B  (one 16-col weight slice)

typedef __bf16 bf16;
typedef bf16  v16bf __attribute__((ext_vector_type(16)));
typedef float v8f   __attribute__((ext_vector_type(8)));

__device__ __forceinline__ unsigned short f2bf(float f) {       // RNE f32->bf16
  union { float f; unsigned u; } x; x.f = f;
  unsigned u = x.u;
  return (unsigned short)((u + 0x7FFFu + ((u >> 16) & 1u)) >> 16);
}
__device__ __forceinline__ float sigmoidf_(float x) { return 1.0f / (1.0f + __expf(-x)); }
__device__ __forceinline__ float tanhf_(float x) {
  float e = __expf(-2.0f * fabsf(x));          // e in (0,1]: no overflow
  return copysignf((1.0f - e) / (1.0f + e), x);
}

// A-fragment (16x32 bf16) per-lane element -> K index (ISA 7.12.2 layout):
// lanes 0-15: V0..3 K=0..7, V4..7 K=16..23 ; lanes 16-31: +8
__device__ __forceinline__ int kA_map(int lane, int e) {
  int j = e >> 1, h = e & 1;
  int k = (j < 4) ? (2 * j + h) : (8 + 2 * j + h);
  return ((lane < 16) ? 0 : 8) + k;
}

// Scatter one f32 value as bf16 into a packed A-fragment buffer at (row b, K=c).
__device__ __forceinline__ void pack_store(unsigned short* dst, int b, int c, float v) {
  int kt = c >> 5, kl = c & 31;
  int half = (kl >> 3) & 1;                                   // which lane-half holds this K
  int e = (kl < 16) ? (kl - 8 * half) : (kl - 8 * half - 8);  // inverse of kA_map
  int mt = b >> 4, ml = b & 15;
  dst[((mt * KT_H + kt) * 32 + half * 16 + ml) * 16 + e] = f2bf(v);
}

__device__ __forceinline__ void grid_barrier(unsigned* ctr, unsigned epoch) {
  __threadfence();
  __syncthreads();
  if (threadIdx.x == 0) {
    atomicAdd(ctr, 1u);
    unsigned tgt = epoch * NWG;   // monotone counter: no sense reversal needed
    while (__hip_atomic_load(ctr, __ATOMIC_ACQUIRE, __HIP_MEMORY_SCOPE_AGENT) < tgt)
      __builtin_amdgcn_s_sleep(1);
  }
  __syncthreads();
}

// ---- pack Wz|Wr|Wh (f32 [L,HX] row-major, y = hx W^T) into bf16 B-fragments.
// B-frag (32x16): lane holds column n = lane&15; K = (lane<16?0:8)+e+(e<8?0:8).
__global__ void k_pack_w(const float* __restrict__ Wz, const float* __restrict__ Wr,
                         const float* __restrict__ Wh, unsigned short* __restrict__ wpack) {
  int idx = blockIdx.x * 256 + threadIdx.x;        // 96 ntiles * 24 kt * 512
  int e = idx & 15;
  int lane = (idx >> 4) & 31;
  int frag = idx >> 9;
  int kt = frag % 24;
  int nt = frag / 24;                              // 0..31 z | 32..63 r | 64..95 h
  const float* W; int ct;
  if (nt < 32)      { W = Wz; ct = nt; }
  else if (nt < 64) { W = Wr; ct = nt - 32; }
  else              { W = Wh; ct = nt - 64; }
  int n = ct * 16 + (lane & 15);
  int k = kt * 32 + ((lane < 16) ? 0 : 8) + e + ((e < 8) ? 0 : 8);
  wpack[idx] = f2bf(W[n * HX_ + k]);
}

// ---- pack x [T,B,I] f32 into bf16 A-fragments (one-shot, reused every step)
__global__ void k_pack_x(const float* __restrict__ x, unsigned short* __restrict__ xpack) {
  int idx = blockIdx.x * 256 + threadIdx.x;        // T*4*8*512 = 8388608
  int e = idx & 15;
  int lane = (idx >> 4) & 31;
  int rest = idx >> 9;
  int ktx = rest & 7;
  int mt = (rest >> 3) & 3;
  int t = rest >> 5;
  int b = mt * 16 + (lane & 15);
  int i = ktx * 32 + kA_map(lane, e);
  xpack[idx] = f2bf(x[((size_t)t * B_ + b) * I_ + i]);
}

// ---- init: h_cur = h0, pack h0 into A-fragments, zero barrier counter
__global__ void k_init(const float* __restrict__ h0, unsigned short* __restrict__ hA,
                       unsigned short* __restrict__ rhA, float* __restrict__ h_cur,
                       unsigned* __restrict__ ctr) {
  int idx = blockIdx.x * 256 + threadIdx.x;        // B*L = 32768
  int c = idx & (L_ - 1);
  int b = idx >> 9;
  float v = h0[idx];
  h_cur[idx] = v;
  rhA[idx] = 0;
  pack_store(hA, b, c, v);
  if (idx == 0) *ctr = 0;
}

// ---- persistent GRU kernel: 16 WGs x 256 threads (8 wave32)
__global__ __launch_bounds__(256) void k_gru(
    const unsigned short* __restrict__ wpack, const unsigned short* __restrict__ xpack,
    unsigned short* __restrict__ hA, unsigned short* __restrict__ rhA,
    float* __restrict__ h_cur, float* __restrict__ z_buf,
    const float* __restrict__ bz, const float* __restrict__ br,
    const float* __restrict__ bh, float* __restrict__ out, unsigned* __restrict__ ctr) {
  extern __shared__ char smem[];                   // 6 * 24576 B = 144KB weight slices
  const int w = blockIdx.x;
  const int tid = threadIdx.x;
  const int lane = tid & 31;
  const int wv = tid >> 5;

  // Stage this WG's weight slices into LDS once; resident for all 512 steps.
  {
    int nts[6] = { w * 4, w * 4 + 1, w * 4 + 2, w * 4 + 3, 64 + w * 2, 65 + w * 2 };
    for (int s = 0; s < 6; ++s) {
      const uint4* g = (const uint4*)(wpack + (size_t)nts[s] * 24 * FRAG_EL);
      uint4* l = (uint4*)(smem + s * SLOT_BYTES);
      #pragma unroll
      for (int i = 0; i < 6; ++i)                  // 1536 uint4 / 256 threads
        l[tid + i * 256] = g[tid + i * 256];
    }
  }
  __syncthreads();

  const int m_tile = wv & 3;                       // B=64 -> 4 row tiles
  const int pp = wv >> 2;                          // tile-pair selector
  const int nt0 = w * 4 + pp * 2;                  // phase-1: 2 gate tiles per wave
  const int slot0 = pp * 2;
  const bool is_r = (w >= 8);                      // WGs 0-7: z gate, 8-15: r gate
  const int nt2 = w * 2 + pp;                      // phase-2: 1 candidate tile per wave
  const int slot2 = 4 + pp;
  const int ncl = lane & 15;
  const int bhalf = (lane < 16) ? 0 : 8;           // C/D layout: M = v + bhalf

  const float* bg = is_r ? br : bz;
  const float bg0 = bg[((nt0) & 31) * 16 + ncl];
  const float bg1 = bg[((nt0 + 1) & 31) * 16 + ncl];
  const float bn  = bh[nt2 * 16 + ncl];

  // per-lane base pointers (hoisted: all hot-loop addressing = base + imm)
  const unsigned short* hAl  = hA  + (size_t)m_tile * KT_H * FRAG_EL + lane * 16;
  const unsigned short* rhAl = rhA + (size_t)m_tile * KT_H * FRAG_EL + lane * 16;
  const char* smB0 = smem + (size_t)slot0 * SLOT_BYTES + lane * 32;        // phase-1 B, tile 0
  const char* smB1 = smB0 + SLOT_BYTES;                                    // phase-1 B, tile 1
  const char* smB2 = smem + (size_t)slot2 * SLOT_BYTES + lane * 32;        // phase-2 B

  unsigned ep = 0;
  for (int t = 0; t < T_; ++t) {
    const unsigned short* xtl =
        xpack + (((size_t)t * 4 + m_tile) * KT_X) * FRAG_EL + lane * 16;

    // ---------- phase 1: z,r = sigmoid([h,x] W^T + b) ----------
    v8f a0 = {}; v8f a1 = {};
    #pragma unroll
    for (int kt = 0; kt < KT_H; ++kt) {            // h part: branch-free, imm offsets
      v16bf av = *(const v16bf*)(hAl + kt * FRAG_EL);
      v16bf b0 = *(const v16bf*)(smB0 + kt * 1024);
      v16bf b1 = *(const v16bf*)(smB1 + kt * 1024);
      a0 = __builtin_amdgcn_wmma_f32_16x16x32_bf16(false, av, false, b0, (short)0, a0, false, false);
      a1 = __builtin_amdgcn_wmma_f32_16x16x32_bf16(false, av, false, b1, (short)0, a1, false, false);
    }
    #pragma unroll
    for (int kt = 0; kt < KT_X; ++kt) {            // x part
      v16bf av = *(const v16bf*)(xtl + kt * FRAG_EL);
      v16bf b0 = *(const v16bf*)(smB0 + (KT_H + kt) * 1024);
      v16bf b1 = *(const v16bf*)(smB1 + (KT_H + kt) * 1024);
      a0 = __builtin_amdgcn_wmma_f32_16x16x32_bf16(false, av, false, b0, (short)0, a0, false, false);
      a1 = __builtin_amdgcn_wmma_f32_16x16x32_bf16(false, av, false, b1, (short)0, a1, false, false);
    }
    {
      const int c0 = ((nt0) & 31) * 16 + ncl;
      const int c1 = c0 + 16;
      if (!is_r) {
        #pragma unroll
        for (int v = 0; v < 8; ++v) {
          int b = m_tile * 16 + v + bhalf;
          z_buf[b * L_ + c0] = sigmoidf_(a0[v] + bg0);
          z_buf[b * L_ + c1] = sigmoidf_(a1[v] + bg1);
        }
      } else {
        #pragma unroll
        for (int v = 0; v < 8; ++v) {
          int b = m_tile * 16 + v + bhalf;
          float rh0 = sigmoidf_(a0[v] + bg0) * h_cur[b * L_ + c0];
          float rh1 = sigmoidf_(a1[v] + bg1) * h_cur[b * L_ + c1];
          pack_store(rhA, b, c0, rh0);     // bf16 A-fragments for phase 2
          pack_store(rhA, b, c1, rh1);
        }
      }
    }
    grid_barrier(ctr, ++ep);

    // ---------- phase 2: h' = (1-z)h + z tanh([r*h,x] Wh^T + bh) ----------
    v8f an = {};
    #pragma unroll
    for (int kt = 0; kt < KT_H; ++kt) {            // r*h part
      v16bf av = *(const v16bf*)(rhAl + kt * FRAG_EL);
      v16bf bv = *(const v16bf*)(smB2 + kt * 1024);
      an = __builtin_amdgcn_wmma_f32_16x16x32_bf16(false, av, false, bv, (short)0, an, false, false);
    }
    #pragma unroll
    for (int kt = 0; kt < KT_X; ++kt) {            // x part
      v16bf av = *(const v16bf*)(xtl + kt * FRAG_EL);
      v16bf bv = *(const v16bf*)(smB2 + (KT_H + kt) * 1024);
      an = __builtin_amdgcn_wmma_f32_16x16x32_bf16(false, av, false, bv, (short)0, an, false, false);
    }
    {
      const int c = nt2 * 16 + ncl;
      #pragma unroll
      for (int v = 0; v < 8; ++v) {
        int b = m_tile * 16 + v + bhalf;
        float th = tanhf_(an[v] + bn);
        float z  = z_buf[b * L_ + c];
        float hp = h_cur[b * L_ + c];
        float hn = (1.0f - z) * hp + z * th;
        out[((size_t)t * B_ + b) * L_ + c] = hn;
        h_cur[b * L_ + c] = hn;
        pack_store(hA, b, c, hn);          // next step's A-fragments
      }
    }
    grid_barrier(ctr, ++ep);
  }
}

extern "C" void kernel_launch(void* const* d_in, const int* in_sizes, int n_in,
                              void* d_out, int out_size, void* d_ws, size_t ws_size,
                              hipStream_t stream) {
  const float* x  = (const float*)d_in[0];
  const float* h0 = (const float*)d_in[1];
  const float* Wz = (const float*)d_in[2];
  const float* bz = (const float*)d_in[3];
  const float* Wr = (const float*)d_in[4];
  const float* br = (const float*)d_in[5];
  const float* Wh = (const float*)d_in[6];
  const float* bh = (const float*)d_in[7];
  float* out = (float*)d_out;

  // workspace layout (~19.6 MB total, all 256B-aligned offsets)
  char* ws = (char*)d_ws;
  unsigned short* wpack = (unsigned short*)(ws);              //  2,359,296 B
  unsigned short* xpack = (unsigned short*)(ws + 2359296);    // 16,777,216 B
  unsigned short* hA    = (unsigned short*)(ws + 19136512);   //     65,536 B
  unsigned short* rhA   = (unsigned short*)(ws + 19202048);   //     65,536 B
  float* h_cur          = (float*)(ws + 19267584);            //    131,072 B
  float* z_buf          = (float*)(ws + 19398656);            //    131,072 B
  unsigned* ctr         = (unsigned*)(ws + 19529728);

  // allow 144KB dynamic LDS (attribute set, not a stream op)
  (void)hipFuncSetAttribute(reinterpret_cast<const void*>(k_gru),
                            hipFuncAttributeMaxDynamicSharedMemorySize, 6 * SLOT_BYTES);

  k_pack_w<<<(96 * 24 * 512) / 256, 256, 0, stream>>>(Wz, Wr, Wh, wpack);
  k_pack_x<<<(T_ * 4 * 8 * 512) / 256, 256, 0, stream>>>(x, xpack);
  k_init <<<(B_ * L_) / 256, 256, 0, stream>>>(h0, hA, rhA, h_cur, ctr);
  k_gru  <<<NWG, 256, 6 * SLOT_BYTES, stream>>>(wpack, xpack, hA, rhA, h_cur, z_buf,
                                                bz, br, bh, out, ctr);
}
